// LinearCrossEntropyLoss_1511828489048
// MI455X (gfx1250) — matrix operand order repaired and checked by reading, not compiled, not run on hardware
//
#include <hip/hip_runtime.h>

// ---------------------------------------------------------------------------
// Fused Linear + CrossEntropy for MI455X (gfx1250, wave32, WMMA).
//   hidden [2,1024,1024] f32, weight [128000,1024] f32, targets [2,1024] i32
//   out: scalar f32 mean CE over non-ignored tokens.
// GEMM runs in fp16 via v_wmma_f32_16x16x32_f16 with f32 accumulation;
// logits are never materialized (online logsumexp per V-chunk).
// A tile is staged into LDS with gfx1250 async global->LDS copies.
// ---------------------------------------------------------------------------

typedef __attribute__((ext_vector_type(16))) _Float16 v16h;
typedef __attribute__((ext_vector_type(8)))  _Float16 v8h;
typedef __attribute__((ext_vector_type(8)))  float    v8f;

namespace {
constexpr int   kIgnore = -100;
constexpr int   kB = 2, kS = 1024, kD = 1024, kV = 128000;
constexpr int   kT = kB * kS;                     // 2048 tokens
constexpr int   kMTile = 16;                      // token rows per block
constexpr int   kWaves = 8;                       // 256 threads / block
constexpr int   kColsPerIter  = kWaves * 16;      // 128 logit cols / iter
constexpr int   kColsPerGroup = 5;                // WMMA tiles sharing one A frag
constexpr int   kNGroups      = 5;
constexpr int   kNIter  = kColsPerGroup * kNGroups;   // 25
constexpr int   kVChunk = kColsPerIter * kNIter;      // 3200 cols / block
constexpr int   kNChunk = kV / kVChunk;               // 40 (exact)
constexpr int   kNTiles = kT / kMTile;                // 128 (exact)
constexpr int   kKSteps = kD / 32;                    // 32
constexpr int   kLdsStride = kD + 8;                  // halfs; 2064B rows -> no bank conflicts
constexpr float kNegInf = -3.0e38f;                   // finite sentinel (NaN-safe merges)
}

// gfx1250 async global->LDS copy, 16B per lane (tracked by ASYNCcnt).
__device__ __forceinline__ void async_copy_b128(void* lds_ptr, const void* gptr) {
  const unsigned lds_off = (unsigned)(size_t)lds_ptr;   // addr[31:0] = LDS offset
  asm volatile("global_load_async_to_lds_b128 %0, %1, off"
               :: "v"(lds_off), "v"((unsigned long long)(size_t)gptr)
               : "memory");
}
__device__ __forceinline__ void wait_asynccnt0() {
  asm volatile("s_wait_asynccnt 0" ::: "memory");
}

// --------------------------- hidden f32 -> f16 -----------------------------
__global__ __launch_bounds__(256) void cvt_hidden_f16(const float* __restrict__ h,
                                                      _Float16* __restrict__ o) {
  const int i = (blockIdx.x * 256 + threadIdx.x) * 8;
  const float4* src = (const float4*)(h + i);
  const float4 f0 = src[0];
  const float4 f1 = src[1];
  v8h r;
  r[0] = (_Float16)f0.x; r[1] = (_Float16)f0.y;
  r[2] = (_Float16)f0.z; r[3] = (_Float16)f0.w;
  r[4] = (_Float16)f1.x; r[5] = (_Float16)f1.y;
  r[6] = (_Float16)f1.z; r[7] = (_Float16)f1.w;
  *(v8h*)(o + i) = r;
}

// ---------------- fused GEMM + per-chunk online logsumexp ------------------
// grid = (kNTiles, kNChunk); token-tiles vary fastest so the 128 blocks of a
// V-chunk run adjacently and share that chunk's 13MB of weight through L2.
__global__ __launch_bounds__(256) void fused_lce(const _Float16* __restrict__ hid,
                                                 const float* __restrict__ w,
                                                 const int* __restrict__ tgt,
                                                 float* __restrict__ m_out,
                                                 float* __restrict__ s_out,
                                                 float* __restrict__ t_out) {
  __shared__ __align__(16) _Float16 Ald[kMTile * kLdsStride];
  __shared__ int   tgt_sh[kMTile];
  __shared__ float mrg[kWaves][kMTile][3];

  const int tile  = blockIdx.x;
  const int chunk = blockIdx.y;
  const int tid   = threadIdx.x;
  const int lane  = tid & 31;
  const int wave  = tid >> 5;
  const int half  = lane >> 4;   // lane half selects K half (A and B)
  const int ln    = lane & 15;   // A: M row; B: N column; D: N column

  // ---- stage A tile (16 x 1024 f16) into padded LDS via async copies ----
  {
    const int row  = tid >> 4;
    const int part = tid & 15;
    const _Float16* src = hid + (size_t)(tile * kMTile + row) * kD + part * 64;
    _Float16*       dst = &Ald[row * kLdsStride + part * 64];
#pragma unroll
    for (int j = 0; j < 8; ++j)
      async_copy_b128(dst + j * 8, src + j * 8);
  }
  if (tid < kMTile) tgt_sh[tid] = tgt[tile * kMTile + tid];
  wait_asynccnt0();
  __syncthreads();

  // per-lane targets for the 8 D rows this lane holds (rows r + 8*half)
  int t8[8];
#pragma unroll
  for (int r = 0; r < 8; ++r) t8[r] = tgt_sh[r + 8 * half];

  float m8[8], s8[8], tl[8];
#pragma unroll
  for (int r = 0; r < 8; ++r) { m8[r] = kNegInf; s8[r] = 0.f; tl[r] = kNegInf; }

  const _Float16* Abase = &Ald[ln * kLdsStride];
  // per-lane weight row pointer: B column ln of this wave's N tile; +16*half K offset
  const float* wbase = w + (size_t)(chunk * kVChunk + wave * 16 + ln) * kD + 16 * half;
  const int colBase0 = chunk * kVChunk + wave * 16 + ln;

  for (int g = 0; g < kNGroups; ++g) {
    // prefetch next group's weight rows (global_prefetch_b8)
    if (g + 1 < kNGroups) {
#pragma unroll
      for (int c = 0; c < kColsPerGroup; ++c)
        __builtin_prefetch(wbase + (size_t)((g + 1) * kColsPerGroup + c) * kColsPerIter * kD, 0, 1);
    }

    v8f acc[kColsPerGroup];
#pragma unroll
    for (int c = 0; c < kColsPerGroup; ++c) acc[c] = (v8f){0.f,0.f,0.f,0.f,0.f,0.f,0.f,0.f};

    for (int kk = 0; kk < kKSteps; ++kk) {
      // A fragment (16x32 f16), contiguous K order (k = 16*half + s): per lane
      // two adjacent 16B runs -> two ds_load_b128, combined with a shuffle.
      const v8h q0 = *(const v8h*)(Abase + kk * 32 + 16 * half);
      const v8h q1 = *(const v8h*)(Abase + kk * 32 + 16 * half + 8);
      const v16h a = __builtin_shufflevector(q0, q1,
          0, 1, 2, 3, 4, 5, 6, 7, 8, 9, 10, 11, 12, 13, 14, 15);
#pragma unroll
      for (int c = 0; c < kColsPerGroup; ++c) {
        const int i = g * kColsPerGroup + c;
        // B fragment (32x16 f16): lane ln = column = weight row, contiguous in K
        const float4* wp4 = (const float4*)(wbase + (size_t)i * kColsPerIter * kD + kk * 32);
        v16h b;
#pragma unroll
        for (int q = 0; q < 4; ++q) {
          const float4 f = wp4[q];
          b[4 * q + 0] = (_Float16)f.x;
          b[4 * q + 1] = (_Float16)f.y;
          b[4 * q + 2] = (_Float16)f.z;
          b[4 * q + 3] = (_Float16)f.w;
        }
        acc[c] = __builtin_amdgcn_wmma_f32_16x16x32_f16(
            /*neg_a=*/false, a, /*neg_b=*/false, b,
            /*c_mod=*/(short)0, acc[c], /*reuse_a=*/false, /*reuse_b=*/false);
      }
    }

    // ---- per-lane online logsumexp update + target-logit capture ----
#pragma unroll
    for (int c = 0; c < kColsPerGroup; ++c) {
      const int i   = g * kColsPerGroup + c;
      const int col = colBase0 + i * kColsPerIter;
#pragma unroll
      for (int r = 0; r < 8; ++r) {
        const float v  = acc[c][r];
        const float nm = fmaxf(m8[r], v);
        s8[r] = s8[r] * __expf(m8[r] - nm) + __expf(v - nm);
        m8[r] = nm;
        if (t8[r] == col) tl[r] = v;
      }
    }
  }

  // ---- merge across the 16 lanes of each half (xor shuffles stay in-group) ----
#pragma unroll
  for (int off = 1; off < 16; off <<= 1) {
#pragma unroll
    for (int r = 0; r < 8; ++r) {
      const float om = __shfl_xor(m8[r], off, 32);
      const float os = __shfl_xor(s8[r], off, 32);
      const float nm = fmaxf(m8[r], om);
      s8[r] = s8[r] * __expf(m8[r] - nm) + os * __expf(om - nm);
      m8[r] = nm;
      tl[r] = fmaxf(tl[r], __shfl_xor(tl[r], off, 32));
    }
  }
  if (ln == 0) {
#pragma unroll
    for (int r = 0; r < 8; ++r) {
      mrg[wave][r + 8 * half][0] = m8[r];
      mrg[wave][r + 8 * half][1] = s8[r];
      mrg[wave][r + 8 * half][2] = tl[r];
    }
  }
  __syncthreads();

  // ---- merge across waves, emit per-(chunk, token) partial ----
  if (tid < kMTile) {
    float M = kNegInf, S = 0.f, T = kNegInf;
#pragma unroll
    for (int wv = 0; wv < kWaves; ++wv) {
      const float m = mrg[wv][tid][0];
      const float s = mrg[wv][tid][1];
      const float t = mrg[wv][tid][2];
      const float nm = fmaxf(M, m);
      S = S * __expf(M - nm) + s * __expf(m - nm);
      M = nm;
      T = fmaxf(T, t);
    }
    const size_t idx = (size_t)chunk * kT + (size_t)tile * kMTile + tid;
    m_out[idx] = M;
    s_out[idx] = S;
    t_out[idx] = T;
  }
}

// ------------------------- final reduction to scalar -----------------------
__global__ __launch_bounds__(256) void finalize_lce(const float* __restrict__ m_in,
                                                    const float* __restrict__ s_in,
                                                    const float* __restrict__ t_in,
                                                    const int* __restrict__ tgt,
                                                    float* __restrict__ out) {
  __shared__ float redl[256];
  __shared__ float redn[256];
  const int tid = threadIdx.x;
  float local = 0.f, cnt = 0.f;
  for (int tok = tid; tok < kT; tok += 256) {
    float M = kNegInf, S = 0.f, T = kNegInf;
    for (int c = 0; c < kNChunk; ++c) {
      const size_t idx = (size_t)c * kT + tok;
      const float m = m_in[idx], s = s_in[idx], t = t_in[idx];
      const float nm = fmaxf(M, m);
      S = S * __expf(M - nm) + s * __expf(m - nm);
      M = nm;
      T = fmaxf(T, t);
    }
    if (tgt[tok] != kIgnore) {
      local += (M + __logf(S)) - T;   // nll = logZ - target_logit
      cnt   += 1.f;
    }
  }
  redl[tid] = local;
  redn[tid] = cnt;
  __syncthreads();
  for (int s = 128; s > 0; s >>= 1) {
    if (tid < s) { redl[tid] += redl[tid + s]; redn[tid] += redn[tid + s]; }
    __syncthreads();
  }
  if (tid == 0) {
    const float n = redn[0];
    out[0] = (n == 0.f) ? redl[0] : redl[0] / fmaxf(n, 1.f);
  }
}

// ---------------------------------------------------------------------------
extern "C" void kernel_launch(void* const* d_in, const int* in_sizes, int n_in,
                              void* d_out, int out_size, void* d_ws, size_t ws_size,
                              hipStream_t stream) {
  (void)in_sizes; (void)n_in; (void)out_size; (void)ws_size;
  const float* hidden  = (const float*)d_in[0];   // [2,1024,1024]
  const float* weight  = (const float*)d_in[1];   // [128000,1024]
  const int*   targets = (const int*)d_in[2];     // [2,1024]
  float*       out     = (float*)d_out;           // scalar

  // workspace layout: fp16 hidden (4MB) + 3 partial arrays (40*2048 f32 each)
  _Float16* hid16 = (_Float16*)d_ws;
  float* ws_m = (float*)((char*)d_ws + (size_t)kT * kD * sizeof(_Float16));
  float* ws_s = ws_m + (size_t)kNChunk * kT;
  float* ws_t = ws_s + (size_t)kNChunk * kT;

  cvt_hidden_f16<<<(kT * kD) / (256 * 8), 256, 0, stream>>>(hidden, hid16);
  fused_lce<<<dim3(kNTiles, kNChunk), 256, 0, stream>>>(hid16, weight, targets,
                                                        ws_m, ws_s, ws_t);
  finalize_lce<<<1, 256, 0, stream>>>(ws_m, ws_s, ws_t, targets, out);
}
